// MultiHeadAttention_29850022707651
// MI455X (gfx1250) — matrix operand order repaired and checked by reading, not compiled, not run on hardware
//
#include <hip/hip_runtime.h>

// ---------------------------------------------------------------------------
// MHA + similarity bonus for MI455X (gfx1250, wave32, WMMA bf16)
//
// Sizes: B=2, SEQ=2048, D=1024, H=16, HD=64, SIM_D=256.
// Compute-bound (~75 GFLOP vs ~85MB traffic) -> everything GEMM-shaped runs
// through v_wmma_f32_16x16x32_bf16.  sim_mat precomputed ONCE (not per head).
// GEMM wave tile 32x64: 12 b128 loads per 8 WMMAs (1.5 loads/WMMA).
// Workspace use: ~91 MB.
// ---------------------------------------------------------------------------

#define SEQ   2048
#define DM    1024
#define NH    16
#define HDIM  64
#define SIMD_ 256
#define BATCH 2

static constexpr float SCALE_QK  = 0.125f;  // 64^-0.5
static constexpr float SIM_BOOST = 0.2f;

typedef __attribute__((ext_vector_type(16))) __bf16 v16bf;
typedef __attribute__((ext_vector_type(8)))  __bf16 v8bf;
typedef __attribute__((ext_vector_type(8)))  float  v8f;

__device__ __forceinline__ v8f zero8() {
  v8f z;
#pragma unroll
  for (int i = 0; i < 8; ++i) z[i] = 0.0f;
  return z;
}

__device__ __forceinline__ v8f wmma_bf16(v16bf a, v16bf b, v8f c) {
  // D = A(16x32) * B(32x16) + C(16x16), fp32 accumulate
  return __builtin_amdgcn_wmma_f32_16x16x32_bf16(false, a, false, b,
                                                 (short)0, c, false, false);
}

// A-operand strip per ISA 7.12.2 (16-bit A 16x32): lane-half g holds
// K = {8g..8g+7} and {16+8g..16+8g+7}.  p = row base + k0 (elements).
__device__ __forceinline__ v16bf load_a_strip(const __bf16* p, int g) {
  v8bf lo = *(const v8bf*)(p + 8 * g);
  v8bf hi = *(const v8bf*)(p + 16 + 8 * g);
  v16bf r;
#pragma unroll
  for (int i = 0; i < 8; ++i) { r[i] = lo[i]; r[i + 8] = hi[i]; }
  return r;
}

// B-operand strip (32x16, K-major per lane): lane-half g holds
// K = {8g..8g+7} and {16+8g..16+8g+7} (same order as A).  p = column-row base + k0.
__device__ __forceinline__ v16bf load_b_strip(const __bf16* p, int g) {
  v8bf lo = *(const v8bf*)(p + 8 * g);
  v8bf hi = *(const v8bf*)(p + 16 + 8 * g);
  v16bf r;
#pragma unroll
  for (int i = 0; i < 8; ++i) { r[i] = lo[i]; r[i + 8] = hi[i]; }
  return r;
}

// ---------------------------------------------------------------------------
// fp32 -> bf16 elementwise
// ---------------------------------------------------------------------------
__global__ __launch_bounds__(256) void cvt_f32_bf16(const float* __restrict__ x,
                                                    __bf16* __restrict__ y, int n) {
  int i = blockIdx.x * blockDim.x + threadIdx.x;
  if (i < n) y[i] = (__bf16)x[i];
}

// ---------------------------------------------------------------------------
// Generic NT GEMM: C[m,n] = scale * sum_k A[m,k]*B[n,k] (+bias)
// Wave tile: 32 x 64 (2 A strips + 4 B strips -> 8 WMMAs per k-step).
// mode 0: bf16 row-major C (ldc)
// mode 1: fp32 row-major C (ldc), + bias[n] if bias != nullptr
// mode 2: bf16 transposed per-batch store  C[b][n][k] (Vt layout [B][DM][SEQ])
// mode 3: bf16 tiled C-layout store: tile=(mtile*ldc + ntile), lane*8 chunk
// ---------------------------------------------------------------------------
__global__ __launch_bounds__(256) void gemm_bf16_nt(
    const __bf16* A, const __bf16* Bm, void* Cout,
    int M, int N, int Kd, int lda, int ldb, int ldc,
    float scale, const float* bias, int mode) {
  int lane = threadIdx.x & 31;
  int wid  = threadIdx.x >> 5;
  int gw   = blockIdx.x * (blockDim.x >> 5) + wid;
  int nt64 = N >> 6;
  int mt   = gw / nt64;                 // 32-row tile index
  int nb   = gw - mt * nt64;
  if (mt >= (M >> 5)) return;           // whole-wave uniform exit (EXEC stays full)
  int mrow = lane & 15, g = lane >> 4;

  const __bf16* arow0 = A + (size_t)(mt * 32 + mrow) * lda;
  const __bf16* arow1 = A + (size_t)(mt * 32 + 16 + mrow) * lda;
  v8f acc[2][4];
#pragma unroll
  for (int i = 0; i < 2; ++i)
#pragma unroll
    for (int t = 0; t < 4; ++t) acc[i][t] = zero8();

  for (int k0 = 0; k0 < Kd; k0 += 32) {
    v16bf a0 = load_a_strip(arow0 + k0, g);
    v16bf a1 = load_a_strip(arow1 + k0, g);
#pragma unroll
    for (int t = 0; t < 4; ++t) {
      const __bf16* brow = Bm + (size_t)(nb * 64 + t * 16 + mrow) * ldb + k0;
      v16bf b = load_b_strip(brow, g);
      acc[0][t] = wmma_bf16(a0, b, acc[0][t]);
      acc[1][t] = wmma_bf16(a1, b, acc[1][t]);
    }
  }

  if (mode == 0) {
    __bf16* C = (__bf16*)Cout;
#pragma unroll
    for (int i = 0; i < 2; ++i)
#pragma unroll
      for (int t = 0; t < 4; ++t) {
        int n = nb * 64 + t * 16 + mrow;
#pragma unroll
        for (int r = 0; r < 8; ++r) {
          int m = mt * 32 + i * 16 + r + 8 * g;
          C[(size_t)m * ldc + n] = (__bf16)(acc[i][t][r] * scale);
        }
      }
  } else if (mode == 1) {
    float* C = (float*)Cout;
#pragma unroll
    for (int i = 0; i < 2; ++i)
#pragma unroll
      for (int t = 0; t < 4; ++t) {
        int n = nb * 64 + t * 16 + mrow;
        float bv = bias ? bias[n] : 0.0f;
#pragma unroll
        for (int r = 0; r < 8; ++r) {
          int m = mt * 32 + i * 16 + r + 8 * g;
          C[(size_t)m * ldc + n] = acc[i][t][r] * scale + bv;
        }
      }
  } else if (mode == 2) {
    __bf16* C = (__bf16*)Cout;  // [BATCH][DM][SEQ]
#pragma unroll
    for (int i = 0; i < 2; ++i)
#pragma unroll
      for (int t = 0; t < 4; ++t) {
        int n = nb * 64 + t * 16 + mrow;
#pragma unroll
        for (int r = 0; r < 8; ++r) {
          int m  = mt * 32 + i * 16 + r + 8 * g;  // global token index
          int bb = m >> 11;                        // /SEQ
          int kk = m & (SEQ - 1);
          C[(size_t)bb * DM * SEQ + (size_t)n * SEQ + kk] =
              (__bf16)(acc[i][t][r] * scale);
        }
      }
  } else {  // mode 3: tiled (ldc = N/16 tiles per row of 16x16 tiles)
    __bf16* C = (__bf16*)Cout;
#pragma unroll
    for (int i = 0; i < 2; ++i)
#pragma unroll
      for (int t = 0; t < 4; ++t) {
        v8bf pv;
#pragma unroll
        for (int r = 0; r < 8; ++r) pv[r] = (__bf16)(acc[i][t][r] * scale);
        *(v8bf*)(C + ((size_t)((mt * 2 + i) * ldc + nb * 4 + t)) * 256 + lane * 8) = pv;
      }
  }
}

// ---------------------------------------------------------------------------
// Row-normalize sim features: y[row,:] = x[row,:] / max(||x||, 1e-12), bf16 out
// One wave per 256-wide row.
// ---------------------------------------------------------------------------
__global__ __launch_bounds__(256) void sim_normalize(const float* __restrict__ x,
                                                     __bf16* __restrict__ y, int rows) {
  int lane = threadIdx.x & 31;
  int row  = blockIdx.x * (blockDim.x >> 5) + (threadIdx.x >> 5);
  if (row >= rows) return;
  const float* p = x + (size_t)row * SIMD_;
  float v[8];
  float ss = 0.0f;
#pragma unroll
  for (int i = 0; i < 8; ++i) { v[i] = p[lane + 32 * i]; ss += v[i] * v[i]; }
#pragma unroll
  for (int off = 1; off < 32; off <<= 1) ss += __shfl_xor(ss, off, 32);
  float inv = rsqrtf(fmaxf(ss, 1e-24f));
  __bf16* q = y + (size_t)row * SIMD_;
#pragma unroll
  for (int i = 0; i < 8; ++i) q[lane + 32 * i] = (__bf16)(v[i] * inv);
}

// ---------------------------------------------------------------------------
// Flash attention: one wave per (b, h, 16-query tile); 64 keys per iteration.
// scores tile = smat(C operand) + Qs @ Kp^T via WMMA; online softmax;
// P staged through LDS (C-layout -> A-strips); O += P @ Vt via WMMA.
// ---------------------------------------------------------------------------
__global__ __launch_bounds__(256) void attn_flash(
    const __bf16* __restrict__ Qs, const __bf16* __restrict__ Kp,
    const __bf16* __restrict__ Vt, const __bf16* __restrict__ smat,
    __bf16* __restrict__ Oo) {
  __shared__ __align__(32) __bf16 plds[8][16 * 64];
  int lane = threadIdx.x & 31;
  int wid  = threadIdx.x >> 5;
  int gw   = blockIdx.x * 8 + wid;          // 0 .. B*NH*(SEQ/16)-1 = 4095
  int qt   = gw & (SEQ / 16 - 1);
  int h    = (gw >> 7) & (NH - 1);
  int b    = gw >> 11;
  int mrow = lane & 15, g = lane >> 4;

  // Q A-strips (16x64 head slice) held in registers for the whole key loop.
  const __bf16* qbase = Qs + ((size_t)(b * SEQ + qt * 16 + mrow)) * DM + h * HDIM;
  v16bf aq0 = load_a_strip(qbase, g);
  v16bf aq1 = load_a_strip(qbase + 32, g);

  const __bf16* kbase = Kp + ((size_t)b * SEQ) * DM + h * HDIM;
  const __bf16* vbase = Vt + ((size_t)b) * DM * SEQ + (size_t)(h * HDIM) * SEQ;
  const __bf16* sbase = smat + ((size_t)(b * (SEQ / 16) + qt)) * (SEQ / 16) * 256;

  float mstat[8], lstat[8];
  v8f o[4];
#pragma unroll
  for (int r = 0; r < 8; ++r) { mstat[r] = -1e30f; lstat[r] = 0.0f; }
#pragma unroll
  for (int t = 0; t < 4; ++t) o[t] = zero8();

  __bf16* pl = plds[wid];

  for (int j = 0; j < SEQ / 64; ++j) {
    // --- scores: C := 0.2*sim_mat tile; then += Qs @ Kp^T (k=64 -> 2 wmma) ---
    v8f s[4];
#pragma unroll
    for (int t = 0; t < 4; ++t) {
      v8bf sv = *(const v8bf*)(sbase + (size_t)(j * 4 + t) * 256 + lane * 8);
#pragma unroll
      for (int r = 0; r < 8; ++r) s[t][r] = (float)sv[r];
    }
#pragma unroll
    for (int t = 0; t < 4; ++t) {
      const __bf16* kb = kbase + (size_t)(j * 64 + t * 16 + mrow) * DM;
      s[t] = wmma_bf16(aq0, load_b_strip(kb, g), s[t]);
      s[t] = wmma_bf16(aq1, load_b_strip(kb + 32, g), s[t]);
    }

    // --- online softmax: row max over 64 cols (4 tiles x 16 lanes) ---
    float mx[8];
#pragma unroll
    for (int r = 0; r < 8; ++r)
      mx[r] = fmaxf(fmaxf(s[0][r], s[1][r]), fmaxf(s[2][r], s[3][r]));
#pragma unroll
    for (int off = 1; off < 16; off <<= 1) {
#pragma unroll
      for (int r = 0; r < 8; ++r) mx[r] = fmaxf(mx[r], __shfl_xor(mx[r], off, 32));
    }

    float rowsum[8];
#pragma unroll
    for (int r = 0; r < 8; ++r) {
      float mnew = fmaxf(mstat[r], mx[r]);
      float corr = __expf(mstat[r] - mnew);
      mstat[r] = mnew;
      lstat[r] *= corr;
#pragma unroll
      for (int t = 0; t < 4; ++t) o[t][r] *= corr;
      float rs = 0.0f;
#pragma unroll
      for (int t = 0; t < 4; ++t) {
        float p = __expf(s[t][r] - mnew);
        s[t][r] = p;
        rs += p;
      }
      rowsum[r] = rs;
    }
#pragma unroll
    for (int off = 1; off < 16; off <<= 1) {
#pragma unroll
      for (int r = 0; r < 8; ++r) rowsum[r] += __shfl_xor(rowsum[r], off, 32);
    }
#pragma unroll
    for (int r = 0; r < 8; ++r) lstat[r] += rowsum[r];

    // --- P (C layout) -> LDS row-major 16x64 bf16, then A-strips for P@V ---
#pragma unroll
    for (int t = 0; t < 4; ++t) {
#pragma unroll
      for (int r = 0; r < 8; ++r)
        pl[(r + 8 * g) * 64 + t * 16 + mrow] = (__bf16)s[t][r];
    }
    // same-wave LDS ops are in-order (DScnt); compiler inserts waits.

#pragma unroll
    for (int ks = 0; ks < 2; ++ks) {
      v16bf ap = load_a_strip(pl + mrow * 64 + ks * 32, g);
#pragma unroll
      for (int nt = 0; nt < 4; ++nt) {
        const __bf16* vb = vbase + (size_t)(nt * 16 + mrow) * SEQ + j * 64 + ks * 32;
        o[nt] = wmma_bf16(ap, load_b_strip(vb, g), o[nt]);
      }
    }
  }

  // --- finalize: O /= l, store bf16 head slice ---
#pragma unroll
  for (int nt = 0; nt < 4; ++nt) {
#pragma unroll
    for (int r = 0; r < 8; ++r) {
      int m = qt * 16 + r + 8 * g;
      float val = o[nt][r] / lstat[r];
      Oo[((size_t)(b * SEQ + m)) * DM + h * HDIM + nt * 16 + mrow] = (__bf16)val;
    }
  }
}

// ---------------------------------------------------------------------------
// Host orchestration
// ---------------------------------------------------------------------------
extern "C" void kernel_launch(void* const* d_in, const int* in_sizes, int n_in,
                              void* d_out, int out_size, void* d_ws, size_t ws_size,
                              hipStream_t stream) {
  const float* query = (const float*)d_in[0];
  const float* key_  = (const float*)d_in[1];
  const float* value = (const float*)d_in[2];
  const float* Wq    = (const float*)d_in[3];
  const float* Wk    = (const float*)d_in[4];
  const float* Wv    = (const float*)d_in[5];
  const float* Wo    = (const float*)d_in[6];
  const float* bo    = (const float*)d_in[7];
  const float* Wsim  = (const float*)d_in[8];

  char* ws = (char*)d_ws;
  size_t off = 0;
  auto alloc = [&](size_t elems, size_t esize) -> void* {
    void* p = ws + off;
    off += (elems * esize + 255) & ~(size_t)255;
    return p;
  };

  const size_t TOK = (size_t)BATCH * SEQ;  // 4096 tokens
  __bf16* qb    = (__bf16*)alloc(TOK * DM, 2);
  __bf16* kb    = (__bf16*)alloc(TOK * DM, 2);
  __bf16* vb    = (__bf16*)alloc(TOK * DM, 2);
  __bf16* Wqb   = (__bf16*)alloc((size_t)DM * DM, 2);
  __bf16* Wkb   = (__bf16*)alloc((size_t)DM * DM, 2);
  __bf16* Wvb   = (__bf16*)alloc((size_t)DM * DM, 2);
  __bf16* Wob   = (__bf16*)alloc((size_t)DM * DM, 2);
  __bf16* Wsimb = (__bf16*)alloc((size_t)SIMD_ * DM, 2);
  __bf16* Qsb   = (__bf16*)alloc(TOK * DM, 2);        // Q * SCALE
  __bf16* Kpb   = (__bf16*)alloc(TOK * DM, 2);
  __bf16* Vtb   = (__bf16*)alloc(TOK * DM, 2);        // [B][DM][SEQ]
  float*  simf  = (float*) alloc(TOK * SIMD_, 4);
  __bf16* simnb = (__bf16*)alloc(TOK * SIMD_, 2);
  __bf16* smatb = (__bf16*)alloc((size_t)BATCH * SEQ * SEQ, 2);  // tiled layout
  __bf16* attno = (__bf16*)alloc(TOK * DM, 2);
  (void)ws_size;

  auto cvt = [&](const float* x, __bf16* y, size_t n) {
    cvt_f32_bf16<<<(int)((n + 255) / 256), 256, 0, stream>>>(x, y, (int)n);
  };
  cvt(query, qb, TOK * DM);
  cvt(key_,  kb, TOK * DM);
  cvt(value, vb, TOK * DM);
  cvt(Wq, Wqb, (size_t)DM * DM);
  cvt(Wk, Wkb, (size_t)DM * DM);
  cvt(Wv, Wvb, (size_t)DM * DM);
  cvt(Wo, Wob, (size_t)DM * DM);
  cvt(Wsim, Wsimb, (size_t)SIMD_ * DM);

  auto gemm = [&](const __bf16* A, const __bf16* Bm, void* C, int M, int N, int Kd,
                  int lda, int ldb, int ldc, float sc, const float* bias, int mode) {
    int waves  = (M / 32) * (N / 64);
    int blocks = (waves + 7) / 8;
    gemm_bf16_nt<<<blocks, 256, 0, stream>>>(A, Bm, C, M, N, Kd, lda, ldb, ldc,
                                             sc, bias, mode);
  };

  // Projections (X @ W^T): scale folded into Q; V stored head-transposed.
  gemm(qb, Wqb, Qsb, 4096, 1024, 1024, 1024, 1024, 1024, SCALE_QK, nullptr, 0);
  gemm(kb, Wkb, Kpb, 4096, 1024, 1024, 1024, 1024, 1024, 1.0f,     nullptr, 0);
  gemm(vb, Wvb, Vtb, 4096, 1024, 1024, 1024, 1024, 0,    1.0f,     nullptr, 2);
  gemm(qb, Wsimb, simf, 4096, 256, 1024, 1024, 1024, 256, 1.0f,    nullptr, 1);

  sim_normalize<<<4096 / 8, 256, 0, stream>>>(simf, simnb, 4096);

  // smat[b] = 0.2 * simn[b] @ simn[b]^T, stored in attention-friendly tiles.
  for (int bb = 0; bb < BATCH; ++bb) {
    const __bf16* sp = simnb + (size_t)bb * SEQ * SIMD_;
    gemm(sp, sp, smatb + (size_t)bb * SEQ * SEQ,
         2048, 2048, 256, 256, 256, /*ldc = ntiles*/ 2048 / 16,
         SIM_BOOST, nullptr, 3);
  }

  // Flash attention: 4096 wave-tiles -> 512 blocks of 8 waves.
  attn_flash<<<(BATCH * NH * (SEQ / 16)) / 8, 256, 0, stream>>>(Qsb, Kpb, Vtb,
                                                                smatb, attno);

  // Output projection + bias, fp32 straight into d_out.
  gemm(attno, Wob, d_out, 4096, 1024, 1024, 1024, 1024, 1024, 1.0f, bo, 1);
  (void)in_sizes; (void)n_in; (void)out_size;
}
